// Encoder_8143257994004
// MI455X (gfx1250) — matrix operand order repaired and checked by reading, not compiled, not run on hardware
//
#include <hip/hip_runtime.h>
#include <hip/hip_bf16.h>

typedef __attribute__((ext_vector_type(16))) _Float16 v16h;
typedef __attribute__((ext_vector_type(8)))  _Float16 v8h;
typedef __attribute__((ext_vector_type(8)))  float    v8f;

// ---------------- elementwise helpers ----------------

__global__ void zero_f32(float* __restrict__ p, int n) {
  int i = blockIdx.x * blockDim.x + threadIdx.x;
  if (i < n) p[i] = 0.0f;
}

__global__ void deg_count(const int* __restrict__ dst, int E, float* __restrict__ deg) {
  int i = blockIdx.x * blockDim.x + threadIdx.x;
  if (i < E) atomicAdd(&deg[dst[i]], 1.0f);
}

// deg includes self-loop (+1); dinv = rsqrt(deg+1) in place
__global__ void deg_to_dinv(float* __restrict__ d, int n) {
  int i = blockIdx.x * blockDim.x + threadIdx.x;
  if (i < n) d[i] = rsqrtf(d[i] + 1.0f);
}

__global__ void f32_to_f16(const float* __restrict__ in, _Float16* __restrict__ out, int n) {
  int i = blockIdx.x * blockDim.x + threadIdx.x;
  if (i < n) out[i] = (_Float16)in[i];
}

// Wt[n*K + k] = W[k*N + n]   (W: [K,N] f32 -> Wt: [N,K] f16)
__global__ void transpose_to_f16(const float* __restrict__ W, _Float16* __restrict__ Wt,
                                 int K, int N) {
  int idx = blockIdx.x * blockDim.x + threadIdx.x;
  if (idx < K * N) {
    int n = idx / K;
    int k = idx - n * K;
    Wt[idx] = (_Float16)W[k * N + n];
  }
}

// ---------------- WMMA GEMM: C[M,N] = A[M,K] @ Bt[N,K]^T ----------------
// One wave per 16x16 output tile. M%16==0, K%32==0, N%16==0 assumed.
// Optional per-output-row scale (dinv[src] factor) and per-column bias.
__global__ void wmma_gemm_f16(const _Float16* __restrict__ A,
                              const _Float16* __restrict__ Bt,
                              float* __restrict__ C,
                              const float* __restrict__ rowscale,
                              const float* __restrict__ bias,
                              int M, int N, int K) {
  int wave = (int)((blockIdx.x * blockDim.x + threadIdx.x) >> 5);
  int lane = (int)(threadIdx.x & 31u);
  int tilesN = N >> 4;
  int tm = wave / tilesN;
  int tn = wave - tm * tilesN;
  if ((tm << 4) >= M) return;           // wave-uniform: EXEC stays all-ones for WMMA

  int half = lane >> 4;                 // 0: lanes 0-15, 1: lanes 16-31
  int l16  = lane & 15;

  const _Float16* Ap = A  + (size_t)((tm << 4) + l16) * (size_t)K;
  const _Float16* Bp = Bt + (size_t)((tn << 4) + l16) * (size_t)K;

  v8f acc = {};
#pragma unroll 4
  for (int k0 = 0; k0 < K; k0 += 32) {
    // A 16x32 f16 layout: lanes 0-15 hold K=[0..8)+[16..24), lanes 16-31 K=[8..16)+[24..32)
    union { v16h v; v8h h[2]; } a;
    a.h[0] = *(const v8h*)(Ap + k0 + half * 8);
    a.h[1] = *(const v8h*)(Ap + k0 + 16 + half * 8);
    // B 32x16 f16 layout: lanes 0-15 hold K=[0..8)+[16..24), lanes 16-31 K=[8..16)+[24..32)
    union { v16h v; v8h h[2]; } b;
    b.h[0] = *(const v8h*)(Bp + k0 + half * 8);
    b.h[1] = *(const v8h*)(Bp + k0 + 16 + half * 8);

    acc = __builtin_amdgcn_wmma_f32_16x16x32_f16(false, a.v, false, b.v,
                                                 (short)0, acc, false, false);
  }

  // C/D layout: VGPR r, lane -> row = tm*16 + half*8 + r, col = tn*16 + l16
  int col = (tn << 4) + l16;
#pragma unroll
  for (int r = 0; r < 8; ++r) {
    int m = (tm << 4) + half * 8 + r;
    float v = acc[r];
    if (rowscale) v *= rowscale[m];
    if (bias)     v += bias[col];
    C[(size_t)m * (size_t)N + col] = v;
  }
}

// ---------------- edge scatter-add (self-loops appended virtually) ----------------
// F = 4 << shift features; each thread handles one float4 of one edge.
__global__ void scatter_add_f(const float* __restrict__ hs,
                              const int* __restrict__ src,
                              const int* __restrict__ dst,
                              int E, int Nn, int shift,
                              float* __restrict__ acc) {
  long long t = (long long)blockIdx.x * blockDim.x + threadIdx.x;
  long long stride = (long long)gridDim.x * blockDim.x;
  long long total = ((long long)(E + Nn)) << shift;
  int mask = (1 << shift) - 1;
  int F = 4 << shift;
  for (; t < total; t += stride) {
    int e  = (int)(t >> shift);
    int fo = ((int)(t & mask)) << 2;
    int s, d;
    if (e < E) { s = src[e]; d = dst[e]; }
    else       { s = e - E;  d = s; }        // self loop
    float4 m = *(const float4*)(hs + (size_t)s * (size_t)F + fo);
    float* ap = acc + (size_t)d * (size_t)F + fo;
    atomicAdd(ap + 0, m.x);
    atomicAdd(ap + 1, m.y);
    atomicAdd(ap + 2, m.z);
    atomicAdd(ap + 3, m.w);
  }
}

// out16 = f16( relu( acc * dinv[node] + bias[f] ) ), F = 1<<shift
__global__ void finalize_relu_f16(const float* __restrict__ acc,
                                  const float* __restrict__ dinv,
                                  const float* __restrict__ bias,
                                  int total, int shift,
                                  _Float16* __restrict__ out16) {
  int i = blockIdx.x * blockDim.x + threadIdx.x;
  if (i < total) {
    int node = i >> shift;
    int f = i & ((1 << shift) - 1);
    float v = acc[i] * dinv[node] + bias[f];
    out16[i] = (_Float16)fmaxf(v, 0.0f);
  }
}

// ---------------- launcher ----------------

extern "C" void kernel_launch(void* const* d_in, const int* in_sizes, int n_in,
                              void* d_out, int out_size, void* d_ws, size_t ws_size,
                              hipStream_t stream) {
  (void)n_in; (void)out_size; (void)ws_size;

  const float* x  = (const float*)d_in[0];
  const int*   ei = (const int*)d_in[1];   // [2, E]
  const float* W1 = (const float*)d_in[2];
  const float* b1 = (const float*)d_in[3];
  const float* W2 = (const float*)d_in[4];
  const float* b2 = (const float*)d_in[5];
  const float* Wl = (const float*)d_in[6];
  const float* bl = (const float*)d_in[7];

  const int Nn = in_sizes[0] / 256;   // 50000 nodes (divisible by 16)
  const int E  = in_sizes[1] / 2;     // 1.6M edges
  const int* src = ei;
  const int* dst = ei + E;

  // workspace carve-out (256-byte aligned pieces)
  char* p = (char*)d_ws;
  auto alloc = [&](size_t bytes) {
    char* r = p;
    p += (bytes + 255) & ~(size_t)255;
    return (void*)r;
  };
  float*    dinv = (float*)   alloc((size_t)Nn * 4);
  float*    hs   = (float*)   alloc((size_t)Nn * 256 * 4);  // messages, reused both layers
  float*    acc  = (float*)   alloc((size_t)Nn * 256 * 4);  // accumulator, reused
  _Float16* a16  = (_Float16*)alloc((size_t)Nn * 256 * 2);  // x_f16, reused as h2_f16
  _Float16* h16  = (_Float16*)alloc((size_t)Nn * 256 * 2);  // h1_f16
  _Float16* w1t  = (_Float16*)alloc((size_t)256 * 256 * 2);
  _Float16* w2t  = (_Float16*)alloc((size_t)256 * 128 * 2);
  _Float16* wlt  = (_Float16*)alloc((size_t)128 * 64 * 2);

  const int T = 256;
  int nx = Nn * 256;   // layer-1 feature count
  int n2 = Nn * 128;   // layer-2 feature count

  // degrees -> dinv (shared by both layers)
  zero_f32   <<<(Nn + T - 1) / T, T, 0, stream>>>(dinv, Nn);
  deg_count  <<<(E  + T - 1) / T, T, 0, stream>>>(dst, E, dinv);
  deg_to_dinv<<<(Nn + T - 1) / T, T, 0, stream>>>(dinv, Nn);

  // precision conversions
  f32_to_f16      <<<(nx + T - 1) / T, T, 0, stream>>>(x, a16, nx);
  transpose_to_f16<<<(256 * 256 + T - 1) / T, T, 0, stream>>>(W1, w1t, 256, 256);
  transpose_to_f16<<<(256 * 128 + T - 1) / T, T, 0, stream>>>(W2, w2t, 256, 128);
  transpose_to_f16<<<(128 * 64  + T - 1) / T, T, 0, stream>>>(Wl, wlt, 128, 64);

  // ---- layer 1: hs = (x @ W1) * dinv[row]; acc = scatter(hs); h1 = relu(acc*dinv + b1)
  {
    int waves  = (Nn / 16) * (256 / 16);
    int blocks = (waves * 32 + T - 1) / T;
    wmma_gemm_f16<<<blocks, T, 0, stream>>>(a16, w1t, hs, dinv, nullptr, Nn, 256, 256);
    zero_f32<<<(nx + T - 1) / T, T, 0, stream>>>(acc, nx);
    scatter_add_f<<<8192, T, 0, stream>>>(hs, src, dst, E, Nn, /*shift=*/6, acc);
    finalize_relu_f16<<<(nx + T - 1) / T, T, 0, stream>>>(acc, dinv, b1, nx, /*shift=*/8, h16);
  }

  // ---- layer 2
  {
    int waves  = (Nn / 16) * (128 / 16);
    int blocks = (waves * 32 + T - 1) / T;
    wmma_gemm_f16<<<blocks, T, 0, stream>>>(h16, w2t, hs, dinv, nullptr, Nn, 128, 256);
    zero_f32<<<(n2 + T - 1) / T, T, 0, stream>>>(acc, n2);
    scatter_add_f<<<8192, T, 0, stream>>>(hs, src, dst, E, Nn, /*shift=*/5, acc);
    finalize_relu_f16<<<(n2 + T - 1) / T, T, 0, stream>>>(acc, dinv, b2, n2, /*shift=*/7, a16);
  }

  // ---- output projection: d_out = h2 @ Wl + bl  (f32)
  {
    int waves  = (Nn / 16) * (64 / 16);
    int blocks = (waves * 32 + T - 1) / T;
    wmma_gemm_f16<<<blocks, T, 0, stream>>>(a16, wlt, (float*)d_out, nullptr, bl, Nn, 64, 128);
  }
}
